// LatentSensorMoE_31705448579974
// MI455X (gfx1250) — matrix operand order repaired and checked by reading, not compiled, not run on hardware
//
#include <hip/hip_runtime.h>
#include <hip/hip_bf16.h>

// ---------------------------------------------------------------------------
// Problem constants
// ---------------------------------------------------------------------------
#define EIN   128
#define GIN   64
#define LAT   16
#define NEXP  8
#define H1    512
#define H2    256
#define H3    128
#define BBD   128     // gate backbone dim
#define GH    128     // gating hidden dim
#define CATK  160     // 144 padded to multiple of 32

typedef __attribute__((ext_vector_type(16))) _Float16 v16h;
typedef __attribute__((ext_vector_type(8)))  _Float16 v8h;
typedef __attribute__((ext_vector_type(8)))  float    v8f;

// ---------------------------------------------------------------------------
// WMMA tile GEMM helper.
//   in : LDS f16, row-major, leading dim ldIn (halves).  16 rows at mBase.
//   wp : global f16 weights packed in fragment order:
//        frag index = (n/16)*(K/32) + (k/32); within frag: lane*16 halves,
//        element m of lane l = B[kTile*32 + 8*(l>>4) + m + (m&8)][nTile*16 + (l&15)]
//   out: LDS f16 row-major, D + bias written per 16x16 tile.
// ---------------------------------------------------------------------------
__device__ __forceinline__ void gemm_wave(const _Float16* __restrict__ in, int ldIn,
                                          const _Float16* __restrict__ wp,
                                          int K, int N,
                                          _Float16* __restrict__ out, int ldOut,
                                          const float* __restrict__ bias,
                                          int mBase, int lane)
{
    const int half  = lane >> 4;
    const int nLane = lane & 15;
    const int m0    = mBase + (lane & 15);      // A-matrix row for this lane
    const int ktc   = K >> 5;

    for (int nB = 0; nB < N; nB += 16) {
        v8f c = {0.f, 0.f, 0.f, 0.f, 0.f, 0.f, 0.f, 0.f};
        const _Float16* wcol = wp + ((size_t)(nB >> 4) * ktc) * 512 + lane * 16;
        for (int kB = 0; kB < K; kB += 32) {
            // A fragment: 16x32 f16, lane holds row m0, Ks {half*8+t | t<8} U {16+half*8+(t-8)}
            v16h a;
            const _Float16* arow = in + (size_t)m0 * ldIn + kB + half * 8;
            #pragma unroll
            for (int t = 0; t < 16; ++t) a[t] = arow[t + (t & 8)];
            // B fragment: contiguous 32B per lane (packed by pre-pass)
            v16h b = *(const v16h*)(wcol + (size_t)(kB >> 5) * 512);
            c = __builtin_amdgcn_wmma_f32_16x16x32_f16(
                    false, a, false, b, (short)0, c, false, false);
        }
        const float bv = bias[nB + nLane];
        #pragma unroll
        for (int j = 0; j < 8; ++j) {
            int m = mBase + j + 8 * half;        // D layout: VGPR j -> M = j + 8*(lane/16)
            out[(size_t)m * ldOut + nB + nLane] = (_Float16)(c[j] + bv);
        }
    }
}

// Row-wise LayerNorm (eps 1e-5, affine) + SiLU, in place on LDS f16.
// Two adjacent lanes cooperate on one row (requires nt == 2*rows active threads).
__device__ __forceinline__ void ln_silu2(_Float16* buf, int ld, int rows, int H,
                                         const float* __restrict__ g,
                                         const float* __restrict__ b,
                                         int tid, int nt)
{
    const int part = tid & 1;
    const int Hh   = H >> 1;
    for (int r = tid >> 1; r < rows; r += nt >> 1) {
        _Float16* row = buf + (size_t)r * ld;
        const int c0 = part * Hh, c1 = c0 + Hh;
        float s = 0.f, s2 = 0.f;
        for (int i = c0; i < c1; ++i) { float v = (float)row[i]; s += v; s2 += v * v; }
        s  += __shfl_xor(s, 1);
        s2 += __shfl_xor(s2, 1);
        float mu  = s / H;
        float inv = rsqrtf(s2 / H - mu * mu + 1e-5f);
        for (int i = c0; i < c1; ++i) {
            float v = ((float)row[i] - mu) * inv * g[i] + b[i];
            row[i] = (_Float16)(v / (1.f + __expf(-v)));
        }
    }
}

__device__ __forceinline__ void silu_ip(_Float16* buf, int ld, int rows, int H,
                                        int tid, int nt)
{
    for (int i = tid; i < rows * H; i += nt) {
        int r = i / H, c = i - r * H;
        float v = (float)buf[(size_t)r * ld + c];
        buf[(size_t)r * ld + c] = (_Float16)(v / (1.f + __expf(-v)));
    }
}

// ---------------------------------------------------------------------------
// Pre-pass kernels
// ---------------------------------------------------------------------------
__global__ void cvt_f32_f16_kernel(const float* __restrict__ src,
                                   _Float16* __restrict__ dst, int n)
{
    for (int i = blockIdx.x * blockDim.x + threadIdx.x; i < n;
         i += gridDim.x * blockDim.x)
        dst[i] = (_Float16)src[i];
}

// Pack f32 [srcK][srcN] (row major) into WMMA-B fragment-order f16, padded to KxN.
__global__ void pack_b_kernel(const float* __restrict__ src, _Float16* __restrict__ dst,
                              int K, int N, int srcK, int srcN)
{
    int total = K * N;
    int ktc = K >> 5;
    for (int i = blockIdx.x * blockDim.x + threadIdx.x; i < total;
         i += gridDim.x * blockDim.x) {
        int frag   = i >> 9;          // /512 halves per fragment
        int within = i & 511;
        int lanei  = within >> 4;     // 0..31
        int m      = within & 15;     // element within lane
        int kt = frag % ktc;
        int nt = frag / ktc;
        int k = kt * 32 + 8 * (lanei >> 4) + m + (m & 8);
        int n = nt * 16 + (lanei & 15);
        float v = (k < srcK && n < srcN) ? src[(size_t)k * srcN + n] : 0.f;
        dst[i] = (_Float16)v;
    }
}

__global__ void gb3pad_kernel(const float* __restrict__ src, float* __restrict__ dst)
{
    int i = threadIdx.x;
    if (i < 16) dst[i] = (i < 8) ? src[i] : 0.f;
}

// ---------------------------------------------------------------------------
// Expert kernel: 64 rows x 1 expert per 128-thread (4-wave) workgroup.
//   bufBig: 536-wide (holds 512/128-wide activations), bufMid: 280-wide.
//   Row strides: 1072B / 560B -> 16B aligned, bank-offset 12 dw per row
//   (co-prime walk over 64 banks across a 16-row A tile).
// ---------------------------------------------------------------------------
#define EROWS 64
#define ELDB  536   // 512 + 24
#define ELDM  280   // 256 + 24

__global__ __launch_bounds__(128) void expert_kernel(
    const _Float16* __restrict__ xf16,
    const _Float16* __restrict__ w1p, const _Float16* __restrict__ w2p,
    const _Float16* __restrict__ w3p, const _Float16* __restrict__ w4f,
    const float* __restrict__ eb1, const float* __restrict__ eg1, const float* __restrict__ ebt1,
    const float* __restrict__ eb2, const float* __restrict__ eg2, const float* __restrict__ ebt2,
    const float* __restrict__ eb3, const float* __restrict__ eg3, const float* __restrict__ ebt3,
    const float* __restrict__ eb4,
    float* __restrict__ expert_out)
{
    __shared__ _Float16 bufBig[EROWS * ELDB];
    __shared__ _Float16 bufMid[EROWS * ELDM];

    const int e       = blockIdx.x & 7;
    const int rowBase = (blockIdx.x >> 3) * EROWS;
    const int tid  = threadIdx.x;
    const int lane = tid & 31;
    const int mB   = (tid >> 5) * 16;

    // stage input rows (f16) into bufMid cols 0..127, 16B vector copies
    for (int i = tid; i < EROWS * (EIN / 8); i += 128) {
        int r = i >> 4, c = (i & 15) * 8;
        *(v8h*)(bufMid + (size_t)r * ELDM + c) =
            *(const v8h*)(xf16 + (size_t)(rowBase + r) * EIN + c);
    }
    __syncthreads();

    // L1: 128 -> 512, LN + SiLU
    gemm_wave(bufMid, ELDM, w1p + (size_t)e * EIN * H1, EIN, H1, bufBig, ELDB,
              eb1 + e * H1, mB, lane);
    __syncthreads();
    ln_silu2(bufBig, ELDB, EROWS, H1, eg1 + e * H1, ebt1 + e * H1, tid, 128);
    __syncthreads();

    // L2: 512 -> 256, LN + SiLU
    gemm_wave(bufBig, ELDB, w2p + (size_t)e * H1 * H2, H1, H2, bufMid, ELDM,
              eb2 + e * H2, mB, lane);
    __syncthreads();
    ln_silu2(bufMid, ELDM, EROWS, H2, eg2 + e * H2, ebt2 + e * H2, tid, 128);
    __syncthreads();

    // L3: 256 -> 128, LN + SiLU
    gemm_wave(bufMid, ELDM, w3p + (size_t)e * H2 * H3, H2, H3, bufBig, ELDB,
              eb3 + e * H3, mB, lane);
    __syncthreads();
    ln_silu2(bufBig, ELDB, EROWS, H3, eg3 + e * H3, ebt3 + e * H3, tid, 128);
    __syncthreads();

    // L4: 128 -> 1 per row; two adjacent lanes split the dot product
    {
        int part = tid & 1;
        int r    = tid >> 1;                 // 0..63, all threads active
        const _Float16* row = bufBig + (size_t)r * ELDB + part * (H3 / 2);
        const _Float16* wv  = w4f + e * H3 + part * (H3 / 2);
        float acc = 0.f;
        for (int i = 0; i < H3 / 2; ++i) acc += (float)row[i] * (float)wv[i];
        acc += __shfl_xor(acc, 1);
        if (part == 0)
            expert_out[(size_t)(rowBase + r) * NEXP + e] = acc + eb4[e];
    }
}

// ---------------------------------------------------------------------------
// Gate kernel: full LatentGateNet, 128 rows per 256-thread (8-wave) workgroup.
// ---------------------------------------------------------------------------
#define GROWS 128
#define GLD   184   // 160 + 24 : rows 16B aligned, 368B = 92 dw (≡28 mod 64)

__global__ __launch_bounds__(256) void gate_kernel(
    const _Float16* __restrict__ gf16,
    const _Float16* __restrict__ bW1p, const float* __restrict__ bb1,
    const float* __restrict__ bg1, const float* __restrict__ bbt1,
    const _Float16* __restrict__ bW2p, const float* __restrict__ bb2,
    const float* __restrict__ bg2, const float* __restrict__ bbt2,
    const _Float16* __restrict__ encW1p, const float* __restrict__ encb1,
    const _Float16* __restrict__ encW2p, const float* __restrict__ encb2,
    const _Float16* __restrict__ gW1p, const float* __restrict__ gb1,
    const float* __restrict__ gg1, const float* __restrict__ gbt1,
    const _Float16* __restrict__ gW2p, const float* __restrict__ gb2,
    const _Float16* __restrict__ gW3p, const float* __restrict__ gb3pad,
    float* __restrict__ out, int B)
{
    __shared__ _Float16 buf0[GROWS * GLD];
    __shared__ _Float16 buf1[GROWS * GLD];

    const int rowBase = blockIdx.x * GROWS;
    const int tid  = threadIdx.x;
    const int lane = tid & 31;
    const int mB   = (tid >> 5) * 16;

    // stage gate features (16B vector copies)
    for (int i = tid; i < GROWS * (GIN / 8); i += 256) {
        int r = i >> 3, c = (i & 7) * 8;
        *(v8h*)(buf0 + (size_t)r * GLD + c) =
            *(const v8h*)(gf16 + (size_t)(rowBase + r) * GIN + c);
    }
    __syncthreads();

    // backbone L1: 64 -> 128, LN + SiLU
    gemm_wave(buf0, GLD, bW1p, GIN, BBD, buf1, GLD, bb1, mB, lane);
    __syncthreads();
    ln_silu2(buf1, GLD, GROWS, BBD, bg1, bbt1, tid, 256);
    __syncthreads();

    // backbone L2: 128 -> 128, LN + SiLU   (h -> buf0 cols 0..127)
    gemm_wave(buf1, GLD, bW2p, BBD, BBD, buf0, GLD, bb2, mB, lane);
    __syncthreads();
    ln_silu2(buf0, GLD, GROWS, BBD, bg2, bbt2, tid, 256);
    __syncthreads();

    // encoder L1: 128 -> 64, SiLU
    gemm_wave(buf0, GLD, encW1p, BBD, BBD / 2, buf1, GLD, encb1, mB, lane);
    __syncthreads();
    silu_ip(buf1, GLD, GROWS, BBD / 2, tid, 256);
    __syncthreads();

    // encoder L2: 64 -> 16  (z -> buf0 cols 128..143, i.e. concat position)
    gemm_wave(buf1, GLD, encW2p, BBD / 2, LAT, buf0 + BBD, GLD, encb2, mB, lane);
    __syncthreads();

    // emit z (f32), zero concat padding cols 144..159
    for (int i = tid; i < GROWS * 16; i += 256) {
        int r = i >> 4, c = i & 15;
        out[(size_t)B + (size_t)(rowBase + r) * LAT + c] = (float)buf0[r * GLD + BBD + c];
        buf0[r * GLD + 144 + c] = (_Float16)0.f;
    }
    __syncthreads();

    // gating L1: 160(pad) -> 128, LN + SiLU
    gemm_wave(buf0, GLD, gW1p, CATK, GH, buf1, GLD, gb1, mB, lane);
    __syncthreads();
    ln_silu2(buf1, GLD, GROWS, GH, gg1, gbt1, tid, 256);
    __syncthreads();

    // gating L2: 128 -> 64, SiLU
    gemm_wave(buf1, GLD, gW2p, GH, GH / 2, buf0, GLD, gb2, mB, lane);
    __syncthreads();
    silu_ip(buf0, GLD, GROWS, GH / 2, tid, 256);
    __syncthreads();

    // gating L3: 64 -> 8 (N padded to 16)
    gemm_wave(buf0, GLD, gW3p, GH / 2, 16, buf1, GLD, gb3pad, mB, lane);
    __syncthreads();

    // logits + softmax -> gates
    if (tid < GROWS) {
        float l[NEXP]; float mx = -1e30f;
        for (int k = 0; k < NEXP; ++k) {
            l[k] = (float)buf1[(size_t)tid * GLD + k];
            mx = fmaxf(mx, l[k]);
        }
        float ex[NEXP]; float s = 0.f;
        for (int k = 0; k < NEXP; ++k) { ex[k] = __expf(l[k] - mx); s += ex[k]; }
        float inv = 1.f / s;
        size_t lbase = (size_t)B * 17 + (size_t)(rowBase + tid) * NEXP;
        size_t gbase = (size_t)B * 25 + (size_t)(rowBase + tid) * NEXP;
        for (int k = 0; k < NEXP; ++k) {
            out[lbase + k] = l[k];
            out[gbase + k] = ex[k] * inv;
        }
    }
}

// ---------------------------------------------------------------------------
// Mixing kernel
// ---------------------------------------------------------------------------
__global__ void mix_kernel(const float* __restrict__ expert_out,
                           const float* __restrict__ gates,
                           float* __restrict__ mixed, int B)
{
    int b = blockIdx.x * blockDim.x + threadIdx.x;
    if (b < B) {
        float s = 0.f;
        for (int e = 0; e < NEXP; ++e)
            s += gates[(size_t)b * NEXP + e] * expert_out[(size_t)b * NEXP + e];
        mixed[b] = s;
    }
}

// ---------------------------------------------------------------------------
// Host launcher
// ---------------------------------------------------------------------------
extern "C" void kernel_launch(void* const* d_in, const int* in_sizes, int n_in,
                              void* d_out, int out_size, void* d_ws, size_t ws_size,
                              hipStream_t stream)
{
    const int B = in_sizes[0] / EIN;   // 32768

    const float* x   = (const float*)d_in[0];
    const float* gx  = (const float*)d_in[1];
    const float* eW1 = (const float*)d_in[2];
    const float* eb1 = (const float*)d_in[3];
    const float* eg1 = (const float*)d_in[4];
    const float* ebt1= (const float*)d_in[5];
    const float* eW2 = (const float*)d_in[6];
    const float* eb2 = (const float*)d_in[7];
    const float* eg2 = (const float*)d_in[8];
    const float* ebt2= (const float*)d_in[9];
    const float* eW3 = (const float*)d_in[10];
    const float* eb3 = (const float*)d_in[11];
    const float* eg3 = (const float*)d_in[12];
    const float* ebt3= (const float*)d_in[13];
    const float* eW4 = (const float*)d_in[14];
    const float* eb4 = (const float*)d_in[15];
    const float* bW1 = (const float*)d_in[16];
    const float* bb1 = (const float*)d_in[17];
    const float* bg1 = (const float*)d_in[18];
    const float* bbt1= (const float*)d_in[19];
    const float* bW2 = (const float*)d_in[20];
    const float* bb2 = (const float*)d_in[21];
    const float* bg2 = (const float*)d_in[22];
    const float* bbt2= (const float*)d_in[23];
    const float* encW1 = (const float*)d_in[24];
    const float* encb1 = (const float*)d_in[25];
    const float* encW2 = (const float*)d_in[26];
    const float* encb2 = (const float*)d_in[27];
    const float* gW1 = (const float*)d_in[28];
    const float* gb1 = (const float*)d_in[29];
    const float* gg1 = (const float*)d_in[30];
    const float* gbt1= (const float*)d_in[31];
    const float* gW2 = (const float*)d_in[32];
    const float* gb2 = (const float*)d_in[33];
    const float* gW3 = (const float*)d_in[34];
    const float* gb3 = (const float*)d_in[35];

    // workspace carve-out (256B aligned)
    char* ws = (char*)d_ws;
    size_t off = 0;
    auto carve = [&](size_t bytes) -> char* {
        char* p = ws + off;
        off = (off + bytes + 255) & ~(size_t)255;
        return p;
    };
    _Float16* xf16   = (_Float16*)carve((size_t)B * EIN * 2);
    _Float16* gf16   = (_Float16*)carve((size_t)B * GIN * 2);
    _Float16* w1p    = (_Float16*)carve((size_t)NEXP * EIN * H1 * 2);
    _Float16* w2p    = (_Float16*)carve((size_t)NEXP * H1 * H2 * 2);
    _Float16* w3p    = (_Float16*)carve((size_t)NEXP * H2 * H3 * 2);
    _Float16* w4f    = (_Float16*)carve((size_t)NEXP * H3 * 2);
    _Float16* bW1p   = (_Float16*)carve((size_t)GIN * BBD * 2);
    _Float16* bW2p   = (_Float16*)carve((size_t)BBD * BBD * 2);
    _Float16* encW1p = (_Float16*)carve((size_t)BBD * (BBD/2) * 2);
    _Float16* encW2p = (_Float16*)carve((size_t)(BBD/2) * LAT * 2);
    _Float16* gW1p   = (_Float16*)carve((size_t)CATK * GH * 2);
    _Float16* gW2p   = (_Float16*)carve((size_t)GH * (GH/2) * 2);
    _Float16* gW3p   = (_Float16*)carve((size_t)(GH/2) * 16 * 2);
    float*    gb3pad = (float*)carve(16 * 4);
    float*    eout   = (float*)carve((size_t)B * NEXP * 4);

    float* out = (float*)d_out;

    // --- pre-pass: converts + fragment packing -----------------------------
    {
        int n = B * EIN;
        cvt_f32_f16_kernel<<<2048, 256, 0, stream>>>(x, xf16, n);
        n = B * GIN;
        cvt_f32_f16_kernel<<<2048, 256, 0, stream>>>(gx, gf16, n);
        cvt_f32_f16_kernel<<<4, 256, 0, stream>>>(eW4, w4f, NEXP * H3);
        gb3pad_kernel<<<1, 32, 0, stream>>>(gb3, gb3pad);
    }
    for (int e = 0; e < NEXP; ++e) {
        pack_b_kernel<<<(EIN*H1 + 255)/256, 256, 0, stream>>>(
            eW1 + (size_t)e * EIN * H1, w1p + (size_t)e * EIN * H1, EIN, H1, EIN, H1);
        pack_b_kernel<<<(H1*H2 + 255)/256, 256, 0, stream>>>(
            eW2 + (size_t)e * H1 * H2, w2p + (size_t)e * H1 * H2, H1, H2, H1, H2);
        pack_b_kernel<<<(H2*H3 + 255)/256, 256, 0, stream>>>(
            eW3 + (size_t)e * H2 * H3, w3p + (size_t)e * H2 * H3, H2, H3, H2, H3);
    }
    pack_b_kernel<<<(GIN*BBD + 255)/256, 256, 0, stream>>>(bW1, bW1p, GIN, BBD, GIN, BBD);
    pack_b_kernel<<<(BBD*BBD + 255)/256, 256, 0, stream>>>(bW2, bW2p, BBD, BBD, BBD, BBD);
    pack_b_kernel<<<(BBD*(BBD/2) + 255)/256, 256, 0, stream>>>(encW1, encW1p, BBD, BBD/2, BBD, BBD/2);
    pack_b_kernel<<<((BBD/2)*LAT + 255)/256, 256, 0, stream>>>(encW2, encW2p, BBD/2, LAT, BBD/2, LAT);
    pack_b_kernel<<<(CATK*GH + 255)/256, 256, 0, stream>>>(gW1, gW1p, CATK, GH, BBD + LAT, GH);
    pack_b_kernel<<<(GH*(GH/2) + 255)/256, 256, 0, stream>>>(gW2, gW2p, GH, GH/2, GH, GH/2);
    pack_b_kernel<<<((GH/2)*16 + 255)/256, 256, 0, stream>>>(gW3, gW3p, GH/2, 16, GH/2, NEXP);

    // --- gate network ------------------------------------------------------
    gate_kernel<<<B / GROWS, 256, 0, stream>>>(
        gf16,
        bW1p, bb1, bg1, bbt1,
        bW2p, bb2, bg2, bbt2,
        encW1p, encb1, encW2p, encb2,
        gW1p, gb1, gg1, gbt1,
        gW2p, gb2, gW3p, gb3pad,
        out, B);

    // --- expert stack ------------------------------------------------------
    expert_kernel<<<(B / EROWS) * NEXP, 128, 0, stream>>>(
        xf16, w1p, w2p, w3p, w4f,
        eb1, eg1, ebt1, eb2, eg2, ebt2, eb3, eg3, ebt3, eb4,
        eout);

    // --- mix ---------------------------------------------------------------
    mix_kernel<<<(B + 255)/256, 256, 0, stream>>>(eout, out + (size_t)B * 25, out, B);
}
